// RaindropLayer_65472481460619
// MI455X (gfx1250) — matrix operand order repaired and checked by reading, not compiled, not run on hardware
//
#include <hip/hip_runtime.h>

// ---------------- problem constants (from reference) ----------------
#define B_      32
#define I_      12
#define V_      128
#define H_      32
#define T_      100
#define NH_     4096           // V*H
#define N3_     12288          // 3*NH
#define CH_     300
#define NCLASS_ 2

typedef __bf16 bf16_t;
typedef __attribute__((ext_vector_type(16))) __bf16 v16bf;
typedef __attribute__((ext_vector_type(8)))  __bf16 v8bf;
typedef __attribute__((ext_vector_type(8)))  float  v8f;
typedef __attribute__((ext_vector_type(4)))  int    v4i;

// ---------------- gfx1250 async global->LDS staging ----------------
#if defined(__has_builtin)
#  if __has_builtin(__builtin_amdgcn_global_load_async_to_lds_b128)
#    define HAS_ASYNC_LDS 1
#  endif
#  if __has_builtin(__builtin_amdgcn_s_wait_asynccnt)
#    define HAS_WAIT_ASYNC 1
#  endif
#endif

typedef __attribute__((address_space(1))) v4i gv4i_t;   // global 16B chunk
typedef __attribute__((address_space(3))) v4i lv4i_t;   // LDS 16B chunk

// one 16-byte chunk, global -> LDS (async on gfx1250; sync VGPR bounce otherwise)
__device__ __forceinline__ void stage_b128(const bf16_t* g, bf16_t* l) {
#ifdef HAS_ASYNC_LDS
    __builtin_amdgcn_global_load_async_to_lds_b128((gv4i_t*)(void*)g,
                                                   (lv4i_t*)(void*)l, 0, 0);
#else
    *(v8bf*)l = *(const v8bf*)g;
#endif
}
__device__ __forceinline__ void wait_stage_tail() {   // allow 1 stage in flight
#ifdef HAS_WAIT_ASYNC
    __builtin_amdgcn_s_wait_asynccnt(1);
#endif
}
__device__ __forceinline__ void wait_stage_all() {
#ifdef HAS_WAIT_ASYNC
    __builtin_amdgcn_s_wait_asynccnt(0);
#endif
}

// ---------------- WMMA helpers (wave32, CDNA5 layouts) ----------------
// A fragment from an LDS k-slice tile (row stride = 32 halfs).
// lane L: row = base + (L&15); halves 0..15 = K kb..kb+15, kb=(L>>4)*16
__device__ __forceinline__ v16bf load_a_frag_lds(const bf16_t* at, int row_base, int lane) {
    int row = row_base + (lane & 15);
    int kb  = (lane >> 4) * 16;
    const bf16_t* p = at + row * 32 + kb;
    v8bf lo = *(const v8bf*)(p);
    v8bf hi = *(const v8bf*)(p + 8);
    return __builtin_shufflevector(lo, hi, 0,1,2,3,4,5,6,7,8,9,10,11,12,13,14,15);
}

// B fragment: 32x16 bf16 (K x N), weight stored row-major [n][k], ldb = K stride.
// lane L: col = base + (L&15); halves 0..15 = K kb..kb+15, kb=(L>>4)*16 -> one contiguous 32B load
__device__ __forceinline__ v16bf load_b_frag(const bf16_t* __restrict__ W, int ldb,
                                             int col_base, int k0, int lane) {
    int col = col_base + (lane & 15);
    int kb  = (lane >> 4) * 16;
    return *(const v16bf*)(W + (long)col * ldb + k0 + kb);
}

__device__ __forceinline__ v8f wmma_bf16(v16bf a, v16bf b, v8f c) {
    return __builtin_amdgcn_wmma_f32_16x16x32_bf16(false, a, false, b, (short)0, c, false, false);
}

__device__ __forceinline__ float sigmoid_f(float x) { return 1.0f / (1.0f + __expf(-x)); }

// ---------------- K0: support = adj / (rowsum + 1e-6) ----------------
__global__ void support_kernel(const float* __restrict__ adj, float* __restrict__ support) {
    int w = blockIdx.x;          // row
    int v = threadIdx.x;         // 128 threads
    float val = adj[w * V_ + v];
    __shared__ float red[V_];
    red[v] = val;
    __syncthreads();
    for (int s = V_ / 2; s > 0; s >>= 1) {
        if (v < s) red[v] += red[v + s];
        __syncthreads();
    }
    support[w * V_ + v] = val / (red[0] + 1e-6f);
}

// ---------------- K1: f32 -> bf16 convert (grid-stride) ----------------
__global__ void cvt_bf16_kernel(const float* __restrict__ src, bf16_t* __restrict__ dst, long n) {
    long i      = (long)blockIdx.x * blockDim.x + threadIdx.x;
    long stride = (long)gridDim.x * blockDim.x;
    for (; i < n; i += stride) dst[i] = (bf16_t)src[i];
}

// ---------------- K2: conv + 2x graph conv -> representations ----------------
__global__ void __launch_bounds__(128)
repr_kernel(const float* __restrict__ x,        // (B,I,V,T)
            const float* __restrict__ support,  // (V,V)
            const float* __restrict__ conv_w,   // (H,I)
            const float* __restrict__ conv_b,   // (H)
            float* __restrict__ repr_out,       // (B,NH,T)
            bf16_t* __restrict__ repr_bf)       // (T*B, NH)
{
    int t = blockIdx.x, b = blockIdx.y;
    __shared__ float xs[I_ * V_];
    __shared__ float xin[H_ * V_];
    __shared__ float x1[H_ * V_];

    for (int idx = threadIdx.x; idx < I_ * V_; idx += 128) {
        int i = idx / V_, v = idx % V_;
        xs[idx] = x[(((long)b * I_ + i) * V_ + v) * T_ + t];
    }
    __syncthreads();

    for (int idx = threadIdx.x; idx < H_ * V_; idx += 128) {
        int h = idx / V_, v = idx % V_;
        float acc = conv_b[h];
        #pragma unroll
        for (int i = 0; i < I_; ++i) acc += conv_w[h * I_ + i] * xs[i * V_ + v];
        xin[idx] = acc;
    }
    __syncthreads();

    for (int idx = threadIdx.x; idx < H_ * V_; idx += 128) {
        int h = idx / V_, w = idx % V_;
        float acc = 0.0f;
        for (int v = 0; v < V_; ++v) acc += xin[h * V_ + v] * support[w * V_ + v];
        x1[idx] = acc;
    }
    __syncthreads();

    for (int idx = threadIdx.x; idx < H_ * V_; idx += 128) {
        int h = idx / V_, w = idx % V_;
        float acc = 0.0f;
        for (int v = 0; v < V_; ++v) acc += x1[h * V_ + v] * support[w * V_ + v];
        repr_out[((long)b * NH_ + idx) * T_ + t] = acc;          // nh = h*V + w == idx
        repr_bf[((long)t * B_ + b) * NH_ + idx]  = (bf16_t)acc;
    }
}

// ---------------- K3: GX = repr @ w_ih^T + b_ih  (bf16 WMMA, async-LDS staged A) ----------------
// M = T*B = 3200, N = 12288, K = 4096. grid = (96, 50), 256 threads = 8 waves.
// Block shares a 64-row A tile via double-buffered async LDS; each wave owns a 16-col strip.
__global__ void __launch_bounds__(256)
gemm_gx_kernel(const bf16_t* __restrict__ A,     // (3200, 4096)
               const bf16_t* __restrict__ W,     // (12288, 4096)
               const float*  __restrict__ bias,  // (12288)
               float*        __restrict__ GX)    // (3200, 12288)
{
    __shared__ __align__(16) bf16_t atile[2][64 * 32];   // 2 x 4 KB

    int tid  = threadIdx.x;
    int lane = tid & 31;
    int wave = tid >> 5;
    int n0 = (blockIdx.x * 8 + wave) * 16;
    int m0 = blockIdx.y * 64;

    // per-thread staging slot: row = tid>>2 (0..63), 16B chunk = tid&3
    int srow   = tid >> 2;
    int schunk = (tid & 3) * 8;                     // half offset within 32-half k-slice
    const bf16_t* sgp = A + (long)(m0 + srow) * NH_ + schunk;
    bf16_t* slp[2] = { &atile[0][srow * 32 + schunk], &atile[1][srow * 32 + schunk] };

    const bf16_t* wrow = W + (long)(n0 + (lane & 15)) * NH_ + (lane >> 4) * 16;

    stage_b128(sgp, slp[0]);                        // k0 = 0 tile
    v8f acc[4] = {};

    for (int k0 = 0; k0 < NH_; k0 += 32) {
        int cur = (k0 >> 5) & 1;
        if (k0 + 32 < NH_) {
            stage_b128(sgp + k0 + 32, slp[cur ^ 1]);
            wait_stage_tail();
        } else {
            wait_stage_all();
        }
        __syncthreads();                             // buf[cur] ready block-wide

        if (k0 + 512 < NH_) __builtin_prefetch(wrow + k0 + 512, 0, 1);  // weight stream
        v16bf bf = load_b_frag(W, NH_, n0, k0, lane);
        const bf16_t* at = atile[cur];
        #pragma unroll
        for (int mt = 0; mt < 4; ++mt) {
            v16bf af = load_a_frag_lds(at, mt * 16, lane);
            acc[mt] = wmma_bf16(af, bf, acc[mt]);
        }
        __syncthreads();                             // buf[cur] free for next stage
    }

    int n = n0 + (lane & 15);
    float bv = bias[n];
    #pragma unroll
    for (int mt = 0; mt < 4; ++mt) {
        #pragma unroll
        for (int r = 0; r < 8; ++r) {
            int m = m0 + mt * 16 + r + 8 * (lane >> 4);
            GX[(long)m * N3_ + n] = acc[mt][r] + bv;
        }
    }
}

// ---------------- K4: h init ----------------
__global__ void init_h_kernel(const float* __restrict__ h0, float* __restrict__ hf,
                              bf16_t* __restrict__ hb) {
    int i = blockIdx.x * blockDim.x + threadIdx.x;
    if (i < B_ * NH_) {
        float v = h0[i % NH_];
        hf[i] = v;
        hb[i] = (bf16_t)v;
    }
}

// ---------------- K5: one fused GRU step (launched T times) ----------------
// gh = h @ w_hh^T (+b_hh) for 3 gate chunks + gate math. M=32 -> 2 row tiles.
// grid = 32 blocks x 256 threads (8 waves); h k-slice shared via async LDS double buffer.
__global__ void __launch_bounds__(256)
gru_step_kernel(const bf16_t* __restrict__ Hb,    // (32,4096) bf16 h_t
                const float*  __restrict__ Hf,    // (32,4096) f32  h_t
                const bf16_t* __restrict__ Whh,   // (12288,4096)
                const float*  __restrict__ bhh,   // (12288)
                const float*  __restrict__ GX,    // (3200,12288)
                float*        __restrict__ HfOut, // (32,4096)
                bf16_t*       __restrict__ HbOut, // (32,4096)
                float*        __restrict__ states,// (B,NH,T) slab in d_out
                int t)
{
    __shared__ __align__(16) bf16_t htile[2][32 * 32];   // 2 x 2 KB

    int tid  = threadIdx.x;
    int lane = tid & 31;
    int wave = tid >> 5;
    int n0 = (blockIdx.x * 8 + wave) * 16;

    // staging: 32 rows x 4 chunks = 128 slots -> threads 0..127
    bool stager = tid < 128;
    int srow    = tid >> 2;
    int schunk  = (tid & 3) * 8;
    const bf16_t* sgp = Hb + (long)srow * NH_ + schunk;
    bf16_t* slp[2] = { &htile[0][srow * 32 + schunk], &htile[1][srow * 32 + schunk] };

    if (stager) stage_b128(sgp, slp[0]);
    v8f aR[2] = {}, aZ[2] = {}, aN[2] = {};

    for (int k0 = 0; k0 < NH_; k0 += 32) {
        int cur = (k0 >> 5) & 1;
        if (k0 + 32 < NH_) {
            if (stager) stage_b128(sgp + k0 + 32, slp[cur ^ 1]);
            wait_stage_tail();
        } else {
            wait_stage_all();
        }
        __syncthreads();

        const bf16_t* ht = htile[cur];
        v16bf a0 = load_a_frag_lds(ht, 0,  lane);
        v16bf a1 = load_a_frag_lds(ht, 16, lane);
        v16bf bR = load_b_frag(Whh, NH_, n0,           k0, lane);
        v16bf bZ = load_b_frag(Whh, NH_, NH_ + n0,     k0, lane);
        v16bf bN = load_b_frag(Whh, NH_, 2 * NH_ + n0, k0, lane);
        aR[0] = wmma_bf16(a0, bR, aR[0]);
        aR[1] = wmma_bf16(a1, bR, aR[1]);
        aZ[0] = wmma_bf16(a0, bZ, aZ[0]);
        aZ[1] = wmma_bf16(a1, bZ, aZ[1]);
        aN[0] = wmma_bf16(a0, bN, aN[0]);
        aN[1] = wmma_bf16(a1, bN, aN[1]);
        __syncthreads();
    }

    int n = n0 + (lane & 15);
    float br = bhh[n], bz = bhh[NH_ + n], bn = bhh[2 * NH_ + n];

    #pragma unroll
    for (int mt = 0; mt < 2; ++mt) {
        #pragma unroll
        for (int r = 0; r < 8; ++r) {
            int m = mt * 16 + r + 8 * (lane >> 4);       // batch row
            const float* gxp = GX + (long)(t * B_ + m) * N3_;
            float rr = sigmoid_f(gxp[n]       + aR[mt][r] + br);
            float zz = sigmoid_f(gxp[NH_ + n] + aZ[mt][r] + bz);
            float nn = tanhf(gxp[2 * NH_ + n] + rr * (aN[mt][r] + bn));
            float hold = Hf[(long)m * NH_ + n];
            float hnew = (1.0f - zz) * nn + zz * hold;
            HfOut[(long)m * NH_ + n] = hnew;
            HbOut[(long)m * NH_ + n] = (bf16_t)hnew;
            states[((long)m * NH_ + n) * T_ + t] = hnew;
        }
    }
}

// ---------------- K6: classifier head ----------------
__global__ void cls_hidden_kernel(const float* __restrict__ h, const float* __restrict__ w1,
                                  const float* __restrict__ b1, float* __restrict__ hid) {
    int idx = blockIdx.x * blockDim.x + threadIdx.x;
    if (idx >= B_ * CH_) return;
    int b = idx / CH_, j = idx % CH_;
    const float* hp = h + (long)b * NH_;
    const float* wp = w1 + (long)j * NH_;
    float acc = b1[j];
    for (int k = 0; k < NH_; ++k) acc += hp[k] * wp[k];
    hid[idx] = acc > 0.0f ? acc : 0.0f;
}

__global__ void cls_out_kernel(const float* __restrict__ hid, const float* __restrict__ w2,
                               const float* __restrict__ b2, float* __restrict__ out) {
    int idx = threadIdx.x;
    if (idx >= B_ * NCLASS_) return;
    int b = idx / NCLASS_, c = idx % NCLASS_;
    float acc = b2[c];
    for (int j = 0; j < CH_; ++j) acc += hid[b * CH_ + j] * w2[c * CH_ + j];
    out[idx] = acc;
}

__global__ void copy_f32_kernel(const float* __restrict__ src, float* __restrict__ dst, int n) {
    int i = blockIdx.x * blockDim.x + threadIdx.x;
    if (i < n) dst[i] = src[i];
}

// ---------------- host ----------------
extern "C" void kernel_launch(void* const* d_in, const int* in_sizes, int n_in,
                              void* d_out, int out_size, void* d_ws, size_t ws_size,
                              hipStream_t stream) {
    (void)in_sizes; (void)n_in; (void)out_size; (void)ws_size;

    const float* x      = (const float*)d_in[0];
    const float* adj    = (const float*)d_in[1];
    /* mask d_in[2] unused by reference */
    const float* h0     = (const float*)d_in[3];
    const float* conv_w = (const float*)d_in[4];
    const float* conv_b = (const float*)d_in[5];
    const float* w_ih   = (const float*)d_in[6];
    const float* w_hh   = (const float*)d_in[7];
    const float* b_ih   = (const float*)d_in[8];
    const float* b_hh   = (const float*)d_in[9];
    const float* cls_w1 = (const float*)d_in[10];
    const float* cls_b1 = (const float*)d_in[11];
    const float* cls_w2 = (const float*)d_in[12];
    const float* cls_b2 = (const float*)d_in[13];

    // d_out slabs (f32)
    float* out_repr  = (float*)d_out;                                  // (B,NH,T)
    float* out_state = out_repr + (size_t)B_ * NH_ * T_;               // (1,B,NH,T)
    float* out_henc  = out_state + (size_t)B_ * NH_ * T_;              // (B,NH)
    float* out_cls   = out_henc + (size_t)B_ * NH_;                    // (B,2)

    // workspace layout
    char* ws = (char*)d_ws;
    size_t off = 0;
    float*  support  = (float*)(ws + off);  off += (size_t)V_ * V_ * 4;            // 64 KB
    bf16_t* wih_bf   = (bf16_t*)(ws + off); off += (size_t)N3_ * NH_ * 2;          // 100.7 MB
    bf16_t* whh_bf   = (bf16_t*)(ws + off); off += (size_t)N3_ * NH_ * 2;          // 100.7 MB
    bf16_t* repr_bf  = (bf16_t*)(ws + off); off += (size_t)T_ * B_ * NH_ * 2;      // 26.2 MB
    float*  GX       = (float*)(ws + off);  off += (size_t)T_ * B_ * N3_ * 4;      // 157.3 MB
    float*  hf0      = (float*)(ws + off);  off += (size_t)B_ * NH_ * 4;
    float*  hf1      = (float*)(ws + off);  off += (size_t)B_ * NH_ * 4;
    bf16_t* hb0      = (bf16_t*)(ws + off); off += (size_t)B_ * NH_ * 2;
    bf16_t* hb1      = (bf16_t*)(ws + off); off += (size_t)B_ * NH_ * 2;
    float*  hid      = (float*)(ws + off);  off += (size_t)B_ * CH_ * 4;
    float*  hfp[2]   = { hf0, hf1 };
    bf16_t* hbp[2]   = { hb0, hb1 };

    // 0) normalized adjacency
    support_kernel<<<V_, V_, 0, stream>>>(adj, support);

    // 1) weight conversions to bf16 (deterministic each call)
    cvt_bf16_kernel<<<4096, 256, 0, stream>>>(w_ih, wih_bf, (long)N3_ * NH_);
    cvt_bf16_kernel<<<4096, 256, 0, stream>>>(w_hh, whh_bf, (long)N3_ * NH_);

    // 2) conv + graph convs -> representations (f32 out + bf16 GEMM operand)
    repr_kernel<<<dim3(T_, B_), 128, 0, stream>>>(x, support, conv_w, conv_b,
                                                  out_repr, repr_bf);

    // 3) hoisted input-gate GEMM: GX = repr @ w_ih^T + b_ih  (3200 x 12288 x 4096)
    gemm_gx_kernel<<<dim3(N3_ / (16 * 8), (T_ * B_) / 64), 256, 0, stream>>>(
        repr_bf, wih_bf, b_ih, GX);

    // 4) h init
    init_h_kernel<<<(B_ * NH_ + 255) / 256, 256, 0, stream>>>(h0, hfp[0], hbp[0]);

    // 5) sequential GRU: 100 fused (gh GEMM + gates) steps, ping-pong h
    for (int t = 0; t < T_; ++t) {
        int in_s = t & 1, out_s = (t + 1) & 1;
        gru_step_kernel<<<NH_ / (16 * 8), 256, 0, stream>>>(
            hbp[in_s], hfp[in_s], whh_bf, b_hh, GX,
            hfp[out_s], hbp[out_s], out_state, t);
    }
    // final h lives in slot (T % 2) == 0
    copy_f32_kernel<<<(B_ * NH_ + 255) / 256, 256, 0, stream>>>(hfp[0], out_henc, B_ * NH_);

    // 6) classifier head
    cls_hidden_kernel<<<(B_ * CH_ + 255) / 256, 256, 0, stream>>>(hfp[0], cls_w1, cls_b1, hid);
    cls_out_kernel<<<1, 64, 0, stream>>>(hid, cls_w2, cls_b2, out_cls);
}
